// SetCriterion_50319836840688
// MI455X (gfx1250) — matrix operand order repaired and checked by reading, not compiled, not run on hardware
//
#include <hip/hip_runtime.h>
#include <hip/hip_bf16.h>
#include <stdint.h>

typedef _Float16 half8 __attribute__((ext_vector_type(8)));
typedef _Float16 v16h  __attribute__((ext_vector_type(16)));
typedef float    v8f   __attribute__((ext_vector_type(8)));

#define N_TRIALS 8192
#define DIM      512
#define INV_T    10.0f
// LDS tile: 16 rows x 512 halves, padded row stride 520 halves (65 x half8 =
// 1040 B) so the 16 B-fragment lanes rotate across all 64 LDS banks.
#define LDS_ROW8 65          // half8 units per padded row
#define TILE_H8  (16 * LDS_ROW8)

union V16 { v16h v; half8 h[2]; };

// ---------------------------------------------------------------------------
// K0: row-normalize fp32 features -> f16, and build per-column code
//     colcode[j] = label (0..2) if valid else 3
// ---------------------------------------------------------------------------
__global__ void k_normalize(const float* __restrict__ x,
                            const int* __restrict__ labels,
                            const int* __restrict__ bad_tr,
                            _Float16* __restrict__ fh,
                            int* __restrict__ colcode) {
  const int row = blockIdx.x;
  const int t   = threadIdx.x;               // 128 threads, 4 elems each
  const float4 v = ((const float4*)(x + (size_t)row * DIM))[t];
  float ss = v.x * v.x + v.y * v.y + v.z * v.z + v.w * v.w;
  __shared__ float red[128];
  red[t] = ss;
  __syncthreads();
  for (int s = 64; s > 0; s >>= 1) {
    if (t < s) red[t] += red[t + s];
    __syncthreads();
  }
  const float rn = rsqrtf(red[0]);
  _Float16* o = fh + (size_t)row * DIM + t * 4;
  o[0] = (_Float16)(v.x * rn);
  o[1] = (_Float16)(v.y * rn);
  o[2] = (_Float16)(v.z * rn);
  o[3] = (_Float16)(v.w * rn);
  if (t == 0) colcode[row] = (bad_tr[row] == 0) ? labels[row] : 3;
}

// ---------------------------------------------------------------------------
// K0b: per-label valid counts (cnt4[0..2]) and total valid (cnt4[3]);
//      zero the global accumulators.
// ---------------------------------------------------------------------------
__global__ void k_counts(const int* __restrict__ labels,
                         const int* __restrict__ bad_tr,
                         int* __restrict__ cnt4,
                         float* __restrict__ acc_total,
                         int* __restrict__ acc_cnt) {
  __shared__ int c[4];
  const int t = threadIdx.x;
  if (t < 4) c[t] = 0;
  __syncthreads();
  for (int i = t; i < N_TRIALS; i += blockDim.x) {
    if (bad_tr[i] == 0) {
      atomicAdd(&c[labels[i]], 1);
      atomicAdd(&c[3], 1);
    }
  }
  __syncthreads();
  if (t < 4) cnt4[t] = c[t];
  if (t == 0) { *acc_total = 0.0f; *acc_cnt = 0; }
}

// ---------------------------------------------------------------------------
// Main: each wave owns 16 anchor rows (A fragments resident in VGPRs).
// Column tiles are staged Global->LDS once per BLOCK with async-to-LDS copies
// (double buffered, ASYNCcnt-tracked) and shared by all 8 waves via ds reads.
// Pass 0: negsum_i = sum over valid different-label j of exp(S_ij)  (logits
//         bounded by ~|1/T| so plain fp32 sum-of-exp is safe -> L = log(sum)).
// Pass 1: possum_i = sum over valid same-label j!=i of log1p(exp(L_i - S_ij)).
// ---------------------------------------------------------------------------
__global__ void __launch_bounds__(256, 1)
k_contrastive(const _Float16* __restrict__ fh,
              const int* __restrict__ colcode,
              const int* __restrict__ cnt4,
              const int* __restrict__ labels,
              const int* __restrict__ bad_tr,
              float* __restrict__ acc_total,
              int* __restrict__ acc_cnt) {
  __shared__ half8 Bs8[2][TILE_H8] __attribute__((aligned(64)));

  const int tid   = threadIdx.x;
  const int lane  = tid & 31;
  const int wave  = tid >> 5;
  const int mbase = blockIdx.x * 128 + wave * 16;
  const int lo16  = lane & 15;
  const int hi    = lane >> 4;               // 0 or 1 (lane half)

  // ---- preload A fragments (rows mbase..mbase+15, all 512 K) ----
  // A 16x32 f16 layout: lane = M row; lanes 0-15 hold K 0-7 & 16-23,
  // lanes 16-31 hold K 8-15 & 24-31 (per K-chunk of 32).
  V16 a[16];
  {
    const _Float16* arow = fh + (size_t)(mbase + lo16) * DIM + hi * 8;
#pragma unroll
    for (int kc = 0; kc < 16; ++kc) {
      a[kc].h[0] = *(const half8*)(arow + kc * 32);
      a[kc].h[1] = *(const half8*)(arow + kc * 32 + 16);
    }
  }

  // D/C layout: lane holds rows rowbase..rowbase+7 (one per acc VGPR),
  // column = tile_n0 + lo16.
  const int rowbase = mbase + hi * 8;
  int alab[8];
#pragma unroll
  for (int r = 0; r < 8; ++r) alab[r] = labels[rowbase + r];

  float negsum[8], poss[8], Lr[8];
#pragma unroll
  for (int r = 0; r < 8; ++r) { negsum[r] = 0.0f; poss[r] = 0.0f; }

  // ---- async tile loader: block cooperatively copies 16 rows x 1 KB ----
  // thread tid copies 64 contiguous bytes of row (tid>>4): 4 async b128 ops.
  const int ldrow  = tid >> 4;               // 0..15
  const int ldcol8 = (tid & 15) * 4;         // half8 index within row (0..60)
  const _Float16* gsrc_base = fh + (size_t)ldrow * DIM + (tid & 15) * 32;

  auto issue_tile = [&](int t0, int buf) {
#pragma unroll
    for (int q = 0; q < 4; ++q) {
      uint64_t ga = (uint64_t)(uintptr_t)(gsrc_base + (size_t)t0 * DIM + q * 8);
      uint32_t la = (uint32_t)(uintptr_t)&Bs8[buf][ldrow * LDS_ROW8 + ldcol8 + q];
      asm volatile("global_load_async_to_lds_b128 %0, %1, off"
                   :: "v"(la), "v"(ga) : "memory");
    }
  };

  const int base8 = lo16 * LDS_ROW8 + hi;     // this lane's B-fragment origin

  for (int pass = 0; pass < 2; ++pass) {
    issue_tile(0, 0);
    asm volatile("s_wait_asynccnt 0x0" ::: "memory");
    __syncthreads();

    for (int t = 0; t < N_TRIALS / 16; ++t) {
      const int buf = t & 1;
      if (t + 1 < N_TRIALS / 16) issue_tile((t + 1) * 16, (t + 1) & 1);

      // ---- compute S tile from LDS-staged B ----
      // B 32x16 f16 layout: lane = N col; lanes 0-15 hold K 0-7 & 16-23,
      // lanes 16-31 hold K 8-15 & 24-31 (per K-chunk of 32).
      v8f c = {};
#pragma unroll
      for (int kc = 0; kc < 16; ++kc) {
        V16 b;
        b.h[0] = Bs8[buf][base8 + kc * 4];
        b.h[1] = Bs8[buf][base8 + kc * 4 + 2];
        c = __builtin_amdgcn_wmma_f32_16x16x32_f16(false, a[kc].v, false, b.v,
                                                   (short)0, c, false, false);
      }

      const int n  = t * 16 + lo16;
      const int jc = colcode[n];
      if (pass == 0) {
#pragma unroll
        for (int r = 0; r < 8; ++r) {
          const float x = c[r] * INV_T;
          if (jc < 3 && jc != alab[r]) negsum[r] += __expf(x);
        }
      } else {
#pragma unroll
        for (int r = 0; r < 8; ++r) {
          const float x = c[r] * INV_T;
          if (jc == alab[r] && n != (rowbase + r))
            poss[r] += log1pf(__expf(Lr[r] - x));   // logaddexp(S,L)-S
        }
      }

      asm volatile("s_wait_asynccnt 0x0" ::: "memory");
      __syncthreads();
    }

    if (pass == 0) {
      // reduce over the 16 columns held by this lane-half (xor 1,2,4,8 stays
      // within each 16-lane group on wave32)
#pragma unroll
      for (int off = 1; off < 16; off <<= 1)
#pragma unroll
        for (int r = 0; r < 8; ++r)
          negsum[r] += __shfl_xor(negsum[r], off, 32);
#pragma unroll
      for (int r = 0; r < 8; ++r)
        Lr[r] = (negsum[r] > 0.0f) ? __logf(negsum[r]) : -1e30f;
    }
  }

#pragma unroll
  for (int off = 1; off < 16; off <<= 1)
#pragma unroll
    for (int r = 0; r < 8; ++r) poss[r] += __shfl_xor(poss[r], off, 32);

  // ======================= per-anchor loss ==============================
  if (lo16 == 0) {
    const int totalValid = cnt4[3];
    float lt = 0.0f;
    int   lc = 0;
#pragma unroll
    for (int r = 0; r < 8; ++r) {
      const int  m       = rowbase + r;
      const int  lab     = alab[r];
      const bool valid   = (bad_tr[m] == 0);
      const int  cv      = cnt4[lab];
      const bool has_neg = (totalValid - cv) > 0;
      const int  pos_cnt = cv - (valid ? 1 : 0);
      const bool has_pos = pos_cnt > 0;
      if (valid && has_neg) {                      // "processed"
        const float per = has_pos
            ? (poss[r] / (float)pos_cnt)
            : log1pf(__expf(Lr[r] - INV_T));        // logaddexp(1/T,L)-1/T
        lt += per;
        lc += has_pos ? 1 : 0;
      }
    }
    atomicAdd(acc_total, lt);
    atomicAdd(acc_cnt, lc);
  }
}

// ---------------------------------------------------------------------------
// Finalize: scalar = total / (1 + count)
// ---------------------------------------------------------------------------
__global__ void k_finalize(const float* __restrict__ acc_total,
                           const int* __restrict__ acc_cnt,
                           float* __restrict__ out) {
  if (threadIdx.x == 0 && blockIdx.x == 0)
    out[0] = *acc_total / (float)(1 + *acc_cnt);
}

extern "C" void kernel_launch(void* const* d_in, const int* in_sizes, int n_in,
                              void* d_out, int out_size, void* d_ws, size_t ws_size,
                              hipStream_t stream) {
  const float* feats  = (const float*)d_in[0];
  const int*   labels = (const int*)d_in[1];
  const int*   bad    = (const int*)d_in[2];
  float*       out    = (float*)d_out;

  char* ws = (char*)d_ws;
  _Float16* fh       = (_Float16*)ws;                               // 8 MB
  int*      colcode  = (int*)(ws + (size_t)N_TRIALS * DIM * 2);     // 32 KB
  int*      cnt4     = (int*)((char*)colcode + N_TRIALS * 4);       // 16 B
  float*    acc_tot  = (float*)((char*)cnt4 + 16);
  int*      acc_cnt  = (int*)((char*)acc_tot + 4);

  k_normalize<<<N_TRIALS, 128, 0, stream>>>(feats, labels, bad, fh, colcode);
  k_counts<<<1, 256, 0, stream>>>(labels, bad, cnt4, acc_tot, acc_cnt);
  k_contrastive<<<N_TRIALS / 128, 256, 0, stream>>>(fh, colcode, cnt4, labels,
                                                    bad, acc_tot, acc_cnt);
  k_finalize<<<1, 64, 0, stream>>>(acc_tot, acc_cnt, out);
}
